// SpatialSparseEncoder2D_40544491274843
// MI455X (gfx1250) — matrix operand order repaired and checked by reading, not compiled, not run on hardware
//
#include <hip/hip_runtime.h>
#include <math.h>

// ---------------------------------------------------------------------------
// Types for CDNA5 WMMA (gfx1250, wave32)
// ---------------------------------------------------------------------------
typedef __attribute__((ext_vector_type(16))) __bf16 v16bf;
typedef __attribute__((ext_vector_type(8)))  __bf16 bf16x8;
typedef __attribute__((ext_vector_type(8)))  float  v8f;

#define NTOK   32768        // B*H*W tokens (2*128*128)
#define HEADS  3
#define DHEAD  32
#define SCALE  0.17677669529663687f   // 32^-0.5
#define NBLK   256
#define TOPK   4

static __device__ __forceinline__ __bf16 f2bf(float f) {
  union { float f; unsigned u; } v; v.f = f;
  unsigned r = v.u + 0x7FFFu + ((v.u >> 16) & 1u);   // round-to-nearest-even
  unsigned short h = (unsigned short)(r >> 16);
  __bf16 o; __builtin_memcpy(&o, &h, 2);
  return o;
}

static __device__ __forceinline__ float bf2f(__bf16 h) {
  unsigned short u; __builtin_memcpy(&u, &h, 2);
  unsigned w = ((unsigned)u) << 16;
  float f; __builtin_memcpy(&f, &w, 4);
  return f;
}

static __device__ __forceinline__ float gelu_exact(float x) {
  return 0.5f * x * (1.0f + erff(x * 0.70710678118654752440f));
}

// ---------------------------------------------------------------------------
// Pack fp32 weight (K,N) row-major into WMMA B-fragment order:
//   Wp[((ntile*(K/32) + chunk)*32 + lane)*16 + e]
//   lane: col = ntile*16 + (lane&15), k = chunk*32 + (lane>=16)*16 + e
// Each lane's fragment becomes one contiguous 32-byte run.
// ---------------------------------------------------------------------------
__global__ void pack_w_kernel(const float* __restrict__ W,
                              __bf16* __restrict__ Wp, int K, int N) {
  const int idx = blockIdx.x * 256 + threadIdx.x;
  if (idx >= K * N) return;
  const int e = idx & 15;
  const int lane = (idx >> 4) & 31;
  const int rest = idx >> 9;
  const int nch = K >> 5;
  const int chunk = rest % nch;
  const int ntile = rest / nch;
  const int col = ntile * 16 + (lane & 15);
  const int k = chunk * 32 + ((lane >> 4) << 4) + e;
  Wp[idx] = f2bf(W[(size_t)k * N + col]);
}

// ---------------------------------------------------------------------------
// bf16 WMMA GEMM: Out = act(A[M,K](bf16) @ Wpacked + bias)
// Block = 128 thr = 4 waves; wave owns a 32x32 patch (2x2 fragments).
// grid = (M/128, ceil(N/32)); M%128==0, N%16==0, K%32==0.
// Optional fp32 and/or bf16 outputs.
// ---------------------------------------------------------------------------
__global__ void gemm_bf16_wmma(const __bf16* __restrict__ A,
                               const __bf16* __restrict__ Wp,
                               const float* __restrict__ bias,
                               float* __restrict__ OutF,
                               __bf16* __restrict__ OutB,
                               int M, int K, int N, int act) {
  const int wave = threadIdx.x >> 5;
  const int lane = threadIdx.x & 31;
  const int m0 = blockIdx.x * 128 + wave * 32;
  const int n0 = blockIdx.y * 32;
  if (m0 >= M) return;                       // wave-uniform
  const bool n2 = (n0 + 16) < N;             // second n-fragment valid?
  const int r16 = lane & 15;
  const int hi = lane >> 4;
  const int koff = hi * 16;
  const int nch = K >> 5;

  v8f acc00 = {0.f,0.f,0.f,0.f,0.f,0.f,0.f,0.f};
  v8f acc01 = acc00, acc10 = acc00, acc11 = acc00;

  const __bf16* a0 = A + (size_t)(m0 + r16) * K;
  const __bf16* a1 = a0 + (size_t)16 * K;
  const __bf16* b0 = Wp + ((size_t)(n0 >> 4) * nch * 32 + lane) * 16;
  const __bf16* b1 = Wp + ((size_t)((n0 >> 4) + 1) * nch * 32 + lane) * 16;

  #pragma unroll 3
  for (int c = 0; c < nch; ++c) {
    const int k0 = c << 5;
    // A fragment: two contiguous 8-element (16B) runs per lane
    bf16x8 a0lo = *(const bf16x8*)(a0 + k0 + koff);
    bf16x8 a0hi = *(const bf16x8*)(a0 + k0 + 8 + koff);
    bf16x8 a1lo = *(const bf16x8*)(a1 + k0 + koff);
    bf16x8 a1hi = *(const bf16x8*)(a1 + k0 + 8 + koff);
    v16bf fa0 = __builtin_shufflevector(a0lo, a0hi, 0,1,2,3,4,5,6,7,8,9,10,11,12,13,14,15);
    v16bf fa1 = __builtin_shufflevector(a1lo, a1hi, 0,1,2,3,4,5,6,7,8,9,10,11,12,13,14,15);
    // B fragments: one contiguous 32B run per lane (pre-packed)
    v16bf fb0 = *(const v16bf*)(b0 + (size_t)c * 512);
    acc00 = __builtin_amdgcn_wmma_f32_16x16x32_bf16(false, fa0, false, fb0, (short)0, acc00, false, false);
    acc10 = __builtin_amdgcn_wmma_f32_16x16x32_bf16(false, fa1, false, fb0, (short)0, acc10, false, false);
    if (n2) {
      v16bf fb1 = *(const v16bf*)(b1 + (size_t)c * 512);
      acc01 = __builtin_amdgcn_wmma_f32_16x16x32_bf16(false, fa0, false, fb1, (short)0, acc01, false, false);
      acc11 = __builtin_amdgcn_wmma_f32_16x16x32_bf16(false, fa1, false, fb1, (short)0, acc11, false, false);
    }
  }

  const int row0 = m0 + hi * 8;              // C/D: vgpr r -> M = r + 8*hi
  #pragma unroll
  for (int t = 0; t < 2; ++t) {
    if (t == 1 && !n2) break;
    const int col = n0 + t * 16 + r16;
    const float bv = bias ? bias[col] : 0.f;
    const v8f accA = t ? acc01 : acc00;
    const v8f accB = t ? acc11 : acc10;
    #pragma unroll
    for (int r = 0; r < 8; ++r) {
      float v0 = accA[r] + bv;
      float v1 = accB[r] + bv;
      if (act) { v0 = gelu_exact(v0); v1 = gelu_exact(v1); }
      const size_t i0 = (size_t)(row0 + r) * N + col;
      const size_t i1 = (size_t)(row0 + 16 + r) * N + col;
      if (OutF) { OutF[i0] = v0; OutF[i1] = v1; }
      if (OutB) { OutB[i0] = f2bf(v0); OutB[i1] = f2bf(v1); }
    }
  }
}

// ---------------------------------------------------------------------------
// Conv k7 s4 p3 (3->96) + channel LayerNorm. One block (96 thr) per pixel.
// Writes fp32 (for residual/gate path) and bf16 (for GEMM consumers).
// ---------------------------------------------------------------------------
__global__ void conv_ln_kernel(const float* __restrict__ x,
                               const float* __restrict__ w,
                               const float* __restrict__ g,
                               const float* __restrict__ bta,
                               float* __restrict__ feat,
                               __bf16* __restrict__ featb) {
  __shared__ float xin[147];
  __shared__ float red[96];
  __shared__ float s_mean, s_rinv;
  const int pix = blockIdx.x;
  const int b = pix >> 14;
  const int rem = pix & 16383;
  const int oy = rem >> 7, ox = rem & 127;
  const int tid = threadIdx.x;

  for (int idx = tid; idx < 147; idx += 96) {
    const int i = idx / 49, r = idx - i * 49;
    const int ky = r / 7, kx = r - ky * 7;
    const int iy = oy * 4 - 3 + ky, ix = ox * 4 - 3 + kx;
    float v = 0.f;
    if (iy >= 0 && iy < 512 && ix >= 0 && ix < 512)
      v = x[((size_t)b * 3 + i) * 262144 + (size_t)iy * 512 + ix];
    xin[idx] = v;
  }
  __syncthreads();

  const float* wr = w + (size_t)tid * 147;
  float acc = 0.f;
  #pragma unroll 7
  for (int idx = 0; idx < 147; ++idx) acc += xin[idx] * wr[idx];
  red[tid] = acc;
  __syncthreads();

  if (tid == 0) {
    float s = 0.f;
    for (int c = 0; c < 96; ++c) s += red[c];
    const float m = s * (1.f / 96.f);
    float s2 = 0.f;
    for (int c = 0; c < 96; ++c) { float d = red[c] - m; s2 += d * d; }
    s_mean = m;
    s_rinv = rsqrtf(s2 * (1.f / 96.f) + 1e-5f);
  }
  __syncthreads();
  const float o = (acc - s_mean) * s_rinv * g[tid] + bta[tid];
  feat[(size_t)pix * 96 + tid] = o;
  featb[(size_t)pix * 96 + tid] = f2bf(o);
}

// ---------------------------------------------------------------------------
// 8x8 block mean -> bf16 (only GEMMs consume it)
// ---------------------------------------------------------------------------
__global__ void block_mean_kernel(const float* __restrict__ feat,
                                  __bf16* __restrict__ bm) {
  const int blk = blockIdx.x;                 // 0..511
  const int b = blk >> 8, bi = blk & 255;
  const int by = bi >> 4, bx = bi & 15;
  const int c = threadIdx.x;
  float s = 0.f;
  #pragma unroll
  for (int py = 0; py < 8; ++py)
    #pragma unroll
    for (int px = 0; px < 8; ++px)
      s += feat[((size_t)b * 16384 + (by * 8 + py) * 128 + (bx * 8 + px)) * 96 + c];
  bm[(size_t)blk * 96 + c] = f2bf(s * (1.f / 64.f));
}

// ---------------------------------------------------------------------------
// scorer head (48 -> 1) + top-4 argmax per batch. One block (256 thr) per b.
// ---------------------------------------------------------------------------
__global__ void score_topk_kernel(const float* __restrict__ h1,
                                  const float* __restrict__ w2,
                                  const float* __restrict__ b2,
                                  int* __restrict__ selIdx) {
  __shared__ float sc[256];
  const int b = blockIdx.x, t = threadIdx.x;
  const float* row = h1 + ((size_t)b * 256 + t) * 48;
  float s = b2[0];
  #pragma unroll
  for (int j = 0; j < 48; ++j) s += row[j] * w2[j];
  sc[t] = s;
  __syncthreads();
  if (t == 0) {
    for (int k = 0; k < TOPK; ++k) {
      int best = 0; float bv = sc[0];
      for (int i = 1; i < 256; ++i)
        if (sc[i] > bv) { bv = sc[i]; best = i; }   // ties -> lowest index
      selIdx[b * TOPK + k] = best;
      sc[best] = -3.4e38f;
    }
  }
}

// ---------------------------------------------------------------------------
// Fused attention over Nk block-level keys (256 cmp / 4 sel), bf16 in/out.
// One thread per query, online softmax. Output (h,d) concat order.
// ---------------------------------------------------------------------------
__global__ void attn_kv_kernel(const __bf16* __restrict__ Qp,
                               const __bf16* __restrict__ Kp,  // (B*256,96)
                               const __bf16* __restrict__ Vp,
                               const int* __restrict__ selIdx, // null -> identity
                               int Nk, __bf16* __restrict__ O) {
  const int q = blockIdx.x * blockDim.x + threadIdx.x;
  if (q >= NTOK) return;
  const int b = q >> 14;
  const __bf16* qrow = Qp + (size_t)q * 96;
  for (int h = 0; h < HEADS; ++h) {
    float qv[DHEAD], accv[DHEAD];
    #pragma unroll
    for (int d = 0; d < DHEAD; ++d) { qv[d] = bf2f(qrow[h * DHEAD + d]); accv[d] = 0.f; }
    float m = -3.4e38f, l = 0.f;
    for (int j = 0; j < Nk; ++j) {
      const int row = b * NBLK + (selIdx ? selIdx[b * TOPK + j] : j);
      const __bf16* kr = Kp + (size_t)row * 96 + h * DHEAD;
      float s = 0.f;
      #pragma unroll
      for (int d = 0; d < DHEAD; ++d) s += qv[d] * bf2f(kr[d]);
      s *= SCALE;
      const float mn = fmaxf(m, s);
      const float corr = __expf(m - mn);
      const float p = __expf(s - mn);
      l = l * corr + p;
      const __bf16* vr = Vp + (size_t)row * 96 + h * DHEAD;
      #pragma unroll
      for (int d = 0; d < DHEAD; ++d) accv[d] = accv[d] * corr + p * bf2f(vr[d]);
      m = mn;
    }
    const float rl = 1.f / l;
    #pragma unroll
    for (int d = 0; d < DHEAD; ++d)
      O[(size_t)q * 96 + h * DHEAD + d] = f2bf(accv[d] * rl);
  }
}

// ---------------------------------------------------------------------------
// Fused 7x7 local-window attention over PROJECTED K/V (projection commutes
// with the gather; edge-pad == coordinate clamp). Output (d,h) interleave.
// ---------------------------------------------------------------------------
__global__ void attn_win_kernel(const __bf16* __restrict__ Qp,
                                const __bf16* __restrict__ Kp,
                                const __bf16* __restrict__ Vp,
                                __bf16* __restrict__ O) {
  const int q = blockIdx.x * blockDim.x + threadIdx.x;
  if (q >= NTOK) return;
  const int b = q >> 14;
  const int rem = q & 16383;
  const int y = rem >> 7, x = rem & 127;
  const __bf16* qrow = Qp + (size_t)q * 96;
  for (int h = 0; h < HEADS; ++h) {
    float qv[DHEAD], accv[DHEAD];
    #pragma unroll
    for (int d = 0; d < DHEAD; ++d) { qv[d] = bf2f(qrow[h * DHEAD + d]); accv[d] = 0.f; }
    float m = -3.4e38f, l = 0.f;
    for (int dy = 0; dy < 7; ++dy) {
      const int ny = min(max(y + dy - 3, 0), 127);
      for (int dx = 0; dx < 7; ++dx) {
        const int nx = min(max(x + dx - 3, 0), 127);
        const int row = b * 16384 + ny * 128 + nx;
        const __bf16* kr = Kp + (size_t)row * 96 + h * DHEAD;
        float s = 0.f;
        #pragma unroll
        for (int d = 0; d < DHEAD; ++d) s += qv[d] * bf2f(kr[d]);
        s *= SCALE;
        const float mn = fmaxf(m, s);
        const float corr = __expf(m - mn);
        const float p = __expf(s - mn);
        l = l * corr + p;
        const __bf16* vr = Vp + (size_t)row * 96 + h * DHEAD;
        #pragma unroll
        for (int d = 0; d < DHEAD; ++d) accv[d] = accv[d] * corr + p * bf2f(vr[d]);
        m = mn;
      }
    }
    const float rl = 1.f / l;
    #pragma unroll
    for (int d = 0; d < DHEAD; ++d)
      O[(size_t)q * 96 + d * HEADS + h] = f2bf(accv[d] * rl);   // (d,h) order
  }
}

// ---------------------------------------------------------------------------
// qn = LN(Q); gate = softmax(qn@gw+gb); Y = w0*yc + w1*ys + w2*yw + Q
// ---------------------------------------------------------------------------
__global__ void gate_combine_kernel(const float* __restrict__ Q,
                                    const float* __restrict__ nqg,
                                    const float* __restrict__ nqb,
                                    const float* __restrict__ gw,
                                    const float* __restrict__ gb,
                                    const float* __restrict__ yc,
                                    const float* __restrict__ ys,
                                    const float* __restrict__ yw,
                                    float* __restrict__ Y,
                                    __bf16* __restrict__ Yb) {
  const int q = blockIdx.x * blockDim.x + threadIdx.x;
  if (q >= NTOK) return;
  const float* qr = Q + (size_t)q * 96;
  float s = 0.f, s2 = 0.f;
  for (int c = 0; c < 96; ++c) { float v = qr[c]; s += v; s2 += v * v; }
  const float mu = s * (1.f / 96.f);
  const float rinv = rsqrtf(s2 * (1.f / 96.f) - mu * mu + 1e-5f);
  float l0 = gb[0], l1 = gb[1], l2 = gb[2];
  for (int c = 0; c < 96; ++c) {
    const float qn = (qr[c] - mu) * rinv * nqg[c] + nqb[c];
    l0 += qn * gw[c * 3 + 0];
    l1 += qn * gw[c * 3 + 1];
    l2 += qn * gw[c * 3 + 2];
  }
  const float mx = fmaxf(l0, fmaxf(l1, l2));
  const float e0 = __expf(l0 - mx), e1 = __expf(l1 - mx), e2 = __expf(l2 - mx);
  const float rs = 1.f / (e0 + e1 + e2);
  const float w0 = e0 * rs, w1 = e1 * rs, w2 = e2 * rs;
  for (int c = 0; c < 96; ++c) {
    const size_t i = (size_t)q * 96 + c;
    const float v = w0 * yc[i] + w1 * ys[i] + w2 * yw[i] + qr[c];
    Y[i] = v;
    Yb[i] = f2bf(v);
  }
}

// ---------------------------------------------------------------------------
// out = LN(Y + FF)
// ---------------------------------------------------------------------------
__global__ void final_ln_kernel(const float* __restrict__ Y,
                                const float* __restrict__ FF,
                                const float* __restrict__ g,
                                const float* __restrict__ bt,
                                float* __restrict__ out) {
  const int q = blockIdx.x * blockDim.x + threadIdx.x;
  if (q >= NTOK) return;
  const size_t base = (size_t)q * 96;
  float s = 0.f, s2 = 0.f;
  for (int c = 0; c < 96; ++c) { float v = Y[base + c] + FF[base + c]; s += v; s2 += v * v; }
  const float mu = s * (1.f / 96.f);
  const float rinv = rsqrtf(s2 * (1.f / 96.f) - mu * mu + 1e-5f);
  for (int c = 0; c < 96; ++c) {
    const float v = Y[base + c] + FF[base + c];
    out[base + c] = (v - mu) * rinv * g[c] + bt[c];
  }
}

// ---------------------------------------------------------------------------
// Host side
// ---------------------------------------------------------------------------
extern "C" void kernel_launch(void* const* d_in, const int* in_sizes, int n_in,
                              void* d_out, int out_size, void* d_ws, size_t ws_size,
                              hipStream_t stream) {
  (void)in_sizes; (void)n_in; (void)out_size; (void)ws_size;
  // jax tree flatten order: params (sorted keys, nested sorted) then x
  const float* cmp_bp = (const float*)d_in[0];
  const float* cmp_wk = (const float*)d_in[1];
  const float* cmp_wp = (const float*)d_in[2];
  const float* cmp_wq = (const float*)d_in[3];
  const float* cmp_wv = (const float*)d_in[4];
  const float* conv_w = (const float*)d_in[5];
  const float* ffn1_b = (const float*)d_in[6];
  const float* ffn1_w = (const float*)d_in[7];
  const float* ffn2_b = (const float*)d_in[8];
  const float* ffn2_w = (const float*)d_in[9];
  const float* gate_b = (const float*)d_in[10];
  const float* gate_w = (const float*)d_in[11];
  const float* mu_b   = (const float*)d_in[12];
  const float* mu_w   = (const float*)d_in[13];
  const float* no_b   = (const float*)d_in[14];
  const float* no_g   = (const float*)d_in[15];
  const float* nq_b   = (const float*)d_in[16];
  const float* nq_g   = (const float*)d_in[17];
  const float* pn_b   = (const float*)d_in[18];
  const float* pn_g   = (const float*)d_in[19];
  const float* sc1_b  = (const float*)d_in[20];
  const float* sc1_w  = (const float*)d_in[21];
  const float* sc2_b  = (const float*)d_in[22];
  const float* sc2_w  = (const float*)d_in[23];
  const float* sel_bp = (const float*)d_in[24];
  const float* sel_wk = (const float*)d_in[25];
  const float* sel_wp = (const float*)d_in[26];
  const float* sel_wq = (const float*)d_in[27];
  const float* sel_wv = (const float*)d_in[28];
  const float* win_bp = (const float*)d_in[29];
  const float* win_wk = (const float*)d_in[30];
  const float* win_wp = (const float*)d_in[31];
  const float* win_wq = (const float*)d_in[32];
  const float* win_wv = (const float*)d_in[33];
  const float* x      = (const float*)d_in[34];

  const size_t TC = (size_t)NTOK * 96;        // 3,145,728 elems
  char* p = (char*)d_ws;
  // fp32 buffers
  float* FEAT = (float*)p; p += TC * 4;
  float* YC   = (float*)p; p += TC * 4;       // reused later as FF2
  float* YS   = (float*)p; p += TC * 4;
  float* YWIN = (float*)p; p += TC * 4;
  float* Y    = (float*)p; p += TC * 4;
  // bf16 buffers
  __bf16* FEATB = (__bf16*)p; p += TC * 2;
  __bf16* QPB   = (__bf16*)p; p += TC * 2;    // per-branch query projection
  __bf16* KWB   = (__bf16*)p; p += TC * 2;    // window K projection
  __bf16* VWB   = (__bf16*)p; p += TC * 2;    // window V projection
  __bf16* OTMPB = (__bf16*)p; p += TC * 2;    // pre-out-proj attention result
  __bf16* YB    = (__bf16*)p; p += TC * 2;
  __bf16* HFFB  = (__bf16*)p; p += TC * 4 * 2; // 32768 x 384 bf16
  // small buffers
  float*  H1  = (float*)p;  p += 512 * 48 * 4;
  __bf16* BMB = (__bf16*)p; p += 512 * 96 * 2;
  __bf16* ZB  = (__bf16*)p; p += 512 * 96 * 2;
  __bf16* KZB = (__bf16*)p; p += 512 * 96 * 2;
  __bf16* VZB = (__bf16*)p; p += 512 * 96 * 2;
  int*    SEL = (int*)p;    p += 256;
  // packed weights (bf16, fragment order)
  __bf16* P_cmp_wq = (__bf16*)p; p += 9216 * 2;
  __bf16* P_cmp_wk = (__bf16*)p; p += 9216 * 2;
  __bf16* P_cmp_wv = (__bf16*)p; p += 9216 * 2;
  __bf16* P_cmp_wp = (__bf16*)p; p += 9216 * 2;
  __bf16* P_sel_wq = (__bf16*)p; p += 9216 * 2;
  __bf16* P_sel_wk = (__bf16*)p; p += 9216 * 2;
  __bf16* P_sel_wv = (__bf16*)p; p += 9216 * 2;
  __bf16* P_sel_wp = (__bf16*)p; p += 9216 * 2;
  __bf16* P_win_wq = (__bf16*)p; p += 9216 * 2;
  __bf16* P_win_wk = (__bf16*)p; p += 9216 * 2;
  __bf16* P_win_wv = (__bf16*)p; p += 9216 * 2;
  __bf16* P_win_wp = (__bf16*)p; p += 9216 * 2;
  __bf16* P_mu    = (__bf16*)p; p += 9216 * 2;
  __bf16* P_sc1   = (__bf16*)p; p += 4608 * 2;
  __bf16* P_ffn1  = (__bf16*)p; p += 36864 * 2;
  __bf16* P_ffn2  = (__bf16*)p; p += 36864 * 2;

  float* FF2  = YC;                           // YC dead before FFN2 writes
  float* Zout = (float*)d_out + TC;           // z output region (B,256,96)
  float* FOUT = (float*)d_out;                // feat output region

  const dim3 blk128(128);
  const dim3 gTok96(NTOK / 128, 96 / 32);     // M=32768, N=96
  const dim3 gTok384(NTOK / 128, 384 / 32);   // M=32768, N=384
  const dim3 gBlk96(512 / 128, 96 / 32);      // M=512,   N=96
  const dim3 gBlk48(512 / 128, 2);            // M=512,   N=48 (half tile guarded)

  // 0) pack all weights into WMMA B-fragment layout (bf16)
  pack_w_kernel<<<36, 256, 0, stream>>>(cmp_wq, P_cmp_wq, 96, 96);
  pack_w_kernel<<<36, 256, 0, stream>>>(cmp_wk, P_cmp_wk, 96, 96);
  pack_w_kernel<<<36, 256, 0, stream>>>(cmp_wv, P_cmp_wv, 96, 96);
  pack_w_kernel<<<36, 256, 0, stream>>>(cmp_wp, P_cmp_wp, 96, 96);
  pack_w_kernel<<<36, 256, 0, stream>>>(sel_wq, P_sel_wq, 96, 96);
  pack_w_kernel<<<36, 256, 0, stream>>>(sel_wk, P_sel_wk, 96, 96);
  pack_w_kernel<<<36, 256, 0, stream>>>(sel_wv, P_sel_wv, 96, 96);
  pack_w_kernel<<<36, 256, 0, stream>>>(sel_wp, P_sel_wp, 96, 96);
  pack_w_kernel<<<36, 256, 0, stream>>>(win_wq, P_win_wq, 96, 96);
  pack_w_kernel<<<36, 256, 0, stream>>>(win_wk, P_win_wk, 96, 96);
  pack_w_kernel<<<36, 256, 0, stream>>>(win_wv, P_win_wv, 96, 96);
  pack_w_kernel<<<36, 256, 0, stream>>>(win_wp, P_win_wp, 96, 96);
  pack_w_kernel<<<36, 256, 0, stream>>>(mu_w,   P_mu,   96, 96);
  pack_w_kernel<<<18, 256, 0, stream>>>(sc1_w,  P_sc1,  96, 48);
  pack_w_kernel<<<144, 256, 0, stream>>>(ffn1_w, P_ffn1, 96, 384);
  pack_w_kernel<<<144, 256, 0, stream>>>(ffn2_w, P_ffn2, 384, 96);

  // 1) conv embed + LayerNorm  (fp32 + bf16)
  conv_ln_kernel<<<NTOK, 96, 0, stream>>>(x, conv_w, pn_g, pn_b, FEAT, FEATB);
  // 2) block means (bf16)
  block_mean_kernel<<<512, 96, 0, stream>>>(FEAT, BMB);
  // 3) z = bm @ mu_w + mu_b  -> fp32 output + bf16 for downstream GEMMs
  gemm_bf16_wmma<<<gBlk96, blk128, 0, stream>>>(BMB, P_mu, mu_b, Zout, ZB, 512, 96, 96, 0);
  // 4) scorer hidden: h1 = gelu(bm @ sc1_w + sc1_b)
  gemm_bf16_wmma<<<gBlk48, blk128, 0, stream>>>(BMB, P_sc1, sc1_b, H1, nullptr, 512, 96, 48, 1);
  // 5) scores + top-4
  score_topk_kernel<<<2, 256, 0, stream>>>(H1, sc2_w, sc2_b, SEL);

  // --- compressed branch ---
  gemm_bf16_wmma<<<gTok96, blk128, 0, stream>>>(FEATB, P_cmp_wq, nullptr, nullptr, QPB, NTOK, 96, 96, 0);
  gemm_bf16_wmma<<<gBlk96, blk128, 0, stream>>>(ZB, P_cmp_wk, nullptr, nullptr, KZB, 512, 96, 96, 0);
  gemm_bf16_wmma<<<gBlk96, blk128, 0, stream>>>(ZB, P_cmp_wv, nullptr, nullptr, VZB, 512, 96, 96, 0);
  attn_kv_kernel<<<NTOK / 256, 256, 0, stream>>>(QPB, KZB, VZB, nullptr, 256, OTMPB);
  gemm_bf16_wmma<<<gTok96, blk128, 0, stream>>>(OTMPB, P_cmp_wp, cmp_bp, YC, nullptr, NTOK, 96, 96, 0);

  // --- selected branch (gather(project(z)) == project(gather(z))) ---
  gemm_bf16_wmma<<<gTok96, blk128, 0, stream>>>(FEATB, P_sel_wq, nullptr, nullptr, QPB, NTOK, 96, 96, 0);
  gemm_bf16_wmma<<<gBlk96, blk128, 0, stream>>>(ZB, P_sel_wk, nullptr, nullptr, KZB, 512, 96, 96, 0);
  gemm_bf16_wmma<<<gBlk96, blk128, 0, stream>>>(ZB, P_sel_wv, nullptr, nullptr, VZB, 512, 96, 96, 0);
  attn_kv_kernel<<<NTOK / 256, 256, 0, stream>>>(QPB, KZB, VZB, SEL, TOPK, OTMPB);
  gemm_bf16_wmma<<<gTok96, blk128, 0, stream>>>(OTMPB, P_sel_wp, sel_bp, YS, nullptr, NTOK, 96, 96, 0);

  // --- window branch (project once, gather in-window) ---
  gemm_bf16_wmma<<<gTok96, blk128, 0, stream>>>(FEATB, P_win_wq, nullptr, nullptr, QPB, NTOK, 96, 96, 0);
  gemm_bf16_wmma<<<gTok96, blk128, 0, stream>>>(FEATB, P_win_wk, nullptr, nullptr, KWB, NTOK, 96, 96, 0);
  gemm_bf16_wmma<<<gTok96, blk128, 0, stream>>>(FEATB, P_win_wv, nullptr, nullptr, VWB, NTOK, 96, 96, 0);
  attn_win_kernel<<<NTOK / 256, 256, 0, stream>>>(QPB, KWB, VWB, OTMPB);
  gemm_bf16_wmma<<<gTok96, blk128, 0, stream>>>(OTMPB, P_win_wp, win_bp, YWIN, nullptr, NTOK, 96, 96, 0);

  // --- gate + residual ---
  gate_combine_kernel<<<NTOK / 256, 256, 0, stream>>>(FEAT, nq_g, nq_b, gate_w, gate_b,
                                                      YC, YS, YWIN, Y, YB);
  // --- FFN (hidden kept bf16-only) ---
  gemm_bf16_wmma<<<gTok384, blk128, 0, stream>>>(YB, P_ffn1, ffn1_b, nullptr, HFFB, NTOK, 96, 384, 1);
  gemm_bf16_wmma<<<gTok96, blk128, 0, stream>>>(HFFB, P_ffn2, ffn2_b, FF2, nullptr, NTOK, 384, 96, 0);
  // --- final LN -> feat output ---
  final_ln_kernel<<<NTOK / 256, 256, 0, stream>>>(Y, FF2, no_g, no_b, FOUT);
}
